// FBSNN_81947976007840
// MI455X (gfx1250) — matrix operand order repaired and checked by reading, not compiled, not run on hardware
//
#include <hip/hip_runtime.h>

#define B_PATHS 4096
#define N_STEPS 50
#define DT      0.02f
#define SQRT_DT 0.14142135623730951f
#define SIGMA_C 0.5f

typedef __attribute__((ext_vector_type(16))) _Float16 v16h;
typedef __attribute__((ext_vector_type(8)))  _Float16 v8h;
typedef __attribute__((ext_vector_type(8)))  float    v8f;

union Frag16 { v16h v; v8h h[2]; };
union HalfPack { _Float16 h[2]; unsigned int u; };

// A-matrix 16x32 f16 fragment: lane (m = L%16, g = L/16)
// halves 0..7  -> k = kh*32 + g*8  + (0..7)
// halves 8..15 -> k = kh*32 + 16 + g*8 + (0..7)
__device__ __forceinline__ v16h ld_fragA(const _Float16* row, int g, int kh) {
  Frag16 f;
  f.h[0] = *(const v8h*)(row + kh * 32 + g * 8);
  f.h[1] = *(const v8h*)(row + kh * 32 + 16 + g * 8);
  return f.v;
}

// B-matrix 32x16 f16 fragment from transposed weights Wt[n][k]:
// lane (n = L%16, g = L/16), halves 0..7 -> k = kh*32 + g*8 + (0..7),
// halves 8..15 -> k = kh*32 + 16 + g*8 + (0..7)
__device__ __forceinline__ v16h ld_fragB(const _Float16* row, int g, int kh) {
  Frag16 f;
  f.h[0] = *(const v8h*)(row + kh * 32 + g * 8);
  f.h[1] = *(const v8h*)(row + kh * 32 + 16 + g * 8);
  return f.v;
}

__device__ __forceinline__ v8f wmma_f16(v16h a, v16h b, v8f c) {
  return __builtin_amdgcn_wmma_f32_16x16x32_f16(false, a, false, b, (short)0, c,
                                                false, false);
}

// input layer: z = t*W0 + y*W1 + bin ; h = sin(z) ; u = cos(z)*W1 (tangent wrt y)
// stores packed as u32 (two f16 per store)
__device__ __forceinline__ void input_layer(float t, float ym,
                                            const float* W0, const float* W1,
                                            const float* bin,
                                            _Float16 (*hb)[64], _Float16 (*ub)[64],
                                            int m, int g) {
  const int j0 = g << 5;
  for (int jp = 0; jp < 16; ++jp) {
    int jj = j0 + 2 * jp;
    HalfPack ph, pu;
#pragma unroll
    for (int e = 0; e < 2; ++e) {
      float z = t * W0[jj + e] + ym * W1[jj + e] + bin[jj + e];
      float s, c;
      __sincosf(z, &s, &c);
      ph.h[e] = (_Float16)s;
      if (ub) pu.h[e] = (_Float16)(c * W1[jj + e]);
    }
    *(unsigned int*)(&hb[m][jj]) = ph.u;
    if (ub) *(unsigned int*)(&ub[m][jj]) = pu.u;
  }
}

// hidden layer, value path only (q-net): h <- sin(h @ W + b)
__device__ __forceinline__ void hidden_h(_Float16 (*hb)[64],
                                         const _Float16 (*Wt)[64],
                                         const float* bias, int m, int g) {
  v16h a0 = ld_fragA(&hb[m][0], g, 0);
  v16h a1 = ld_fragA(&hb[m][0], g, 1);
  v8f acc[4];
#pragma unroll
  for (int tl = 0; tl < 4; ++tl) {
    v16h b0 = ld_fragB(&Wt[tl * 16 + m][0], g, 0);
    v16h b1 = ld_fragB(&Wt[tl * 16 + m][0], g, 1);
    v8f c = {};
    c = wmma_f16(a0, b0, c);
    c = wmma_f16(a1, b1, c);
    acc[tl] = c;
  }
#pragma unroll
  for (int tl = 0; tl < 4; ++tl)
#pragma unroll
    for (int r = 0; r < 8; ++r) {
      int mm = r + (g << 3), nn = tl * 16 + m;
      hb[mm][nn] = (_Float16)__sinf(acc[tl][r] + bias[nn]);
    }
}

// hidden layer with forward-mode tangent (Y-net):
// z = h@W + b ; h <- sin(z) ; u <- cos(z) * (u@W)
__device__ __forceinline__ void hidden_hu(_Float16 (*hb)[64], _Float16 (*ub)[64],
                                          const _Float16 (*Wt)[64],
                                          const float* bias, int m, int g) {
  v16h ah0 = ld_fragA(&hb[m][0], g, 0);
  v16h ah1 = ld_fragA(&hb[m][0], g, 1);
  v16h au0 = ld_fragA(&ub[m][0], g, 0);
  v16h au1 = ld_fragA(&ub[m][0], g, 1);
  v8f accz[4], accu[4];
#pragma unroll
  for (int tl = 0; tl < 4; ++tl) {
    v16h b0 = ld_fragB(&Wt[tl * 16 + m][0], g, 0);
    v16h b1 = ld_fragB(&Wt[tl * 16 + m][0], g, 1);
    v8f cz = {}, cu = {};
    cz = wmma_f16(ah0, b0, cz);
    cz = wmma_f16(ah1, b1, cz);
    cu = wmma_f16(au0, b0, cu);
    cu = wmma_f16(au1, b1, cu);
    accz[tl] = cz;
    accu[tl] = cu;
  }
#pragma unroll
  for (int tl = 0; tl < 4; ++tl)
#pragma unroll
    for (int r = 0; r < 8; ++r) {
      int mm = r + (g << 3), nn = tl * 16 + m;
      float s, c;
      __sincosf(accz[tl][r] + bias[nn], &s, &c);
      hb[mm][nn] = (_Float16)s;
      ub[mm][nn] = (_Float16)(c * accu[tl][r]);
    }
}

// output dots: Y = h . Wout (and dY = u . Wout); rows read as b128 vectors.
// Result valid in all lanes for row m = L%16.
__device__ __forceinline__ float2 out_dots(const _Float16 (*hb)[64],
                                           const _Float16 (*ub)[64],
                                           const float* Wo, int m, int g) {
  float sh = 0.f, su = 0.f;
  const int k0 = g << 5;
#pragma unroll
  for (int kb = 0; kb < 4; ++kb) {
    v8h hv = *(const v8h*)&hb[m][k0 + kb * 8];
    v8h uv = {};
    if (ub) uv = *(const v8h*)&ub[m][k0 + kb * 8];
#pragma unroll
    for (int i = 0; i < 8; ++i) {
      float w = Wo[k0 + kb * 8 + i];
      sh += (float)hv[i] * w;
      if (ub) su += (float)uv[i] * w;
    }
  }
  sh += __shfl_xor(sh, 16);
  if (ub) su += __shfl_xor(su, 16);
  float2 r;
  r.x = sh;
  r.y = su;
  return r;
}

__global__ __launch_bounds__(64) void fbsnn_kernel(
    const float* __restrict__ Y_Win, const float* __restrict__ Y_bin,
    const float* __restrict__ Y_Whid, const float* __restrict__ Y_bhid,
    const float* __restrict__ Y_Wout, const float* __restrict__ Y_bout,
    const float* __restrict__ q_Win, const float* __restrict__ q_bin,
    const float* __restrict__ q_Whid, const float* __restrict__ q_bhid,
    const float* __restrict__ q_Wout, const float* __restrict__ q_bout,
    const float* __restrict__ y0_param, const float* __restrict__ dW,
    float* __restrict__ partials) {
  __shared__ __attribute__((aligned(16))) _Float16 sYW[3][64][64];  // Wt[n][k]
  __shared__ __attribute__((aligned(16))) _Float16 sQW[3][64][64];
  __shared__ float sYb[3][64], sQb[3][64];
  __shared__ float sYW0[64], sYW1[64], sYbi[64];
  __shared__ float sQW0[64], sQW1[64], sQbi[64];
  __shared__ float sYWo[64], sQWo[64];
  __shared__ __attribute__((aligned(16))) _Float16 hbuf[2][16][64];
  __shared__ __attribute__((aligned(16))) _Float16 ubuf[2][16][64];
  __shared__ float ystage[2][16];
  __shared__ float wsum[2];

  const int tid = threadIdx.x;
  const int wv = tid >> 5, lane = tid & 31;
  const int m = lane & 15, g = lane >> 4;

  // Stage weights to LDS: hidden weights transposed to Wt[n][k] in f16.
  for (int d = tid; d < 3 * 64 * 64; d += 64) {
    int l = d >> 12, n = (d >> 6) & 63, k = d & 63;
    sYW[l][n][k] = (_Float16)Y_Whid[(l << 12) + (k << 6) + n];
    sQW[l][n][k] = (_Float16)q_Whid[(l << 12) + (k << 6) + n];
  }
  for (int d = tid; d < 64; d += 64) {
    sYW0[d] = Y_Win[d];      sYW1[d] = Y_Win[64 + d];  sYbi[d] = Y_bin[d];
    sQW0[d] = q_Win[d];      sQW1[d] = q_Win[64 + d];  sQbi[d] = q_bin[d];
    sYWo[d] = Y_Wout[d];     sQWo[d] = q_Wout[d];
    for (int l = 0; l < 3; ++l) {
      sYb[l][d] = Y_bhid[l * 64 + d];
      sQb[l][d] = q_bhid[l * 64 + d];
    }
  }
  const float Ybo = Y_bout[0], qbo = q_bout[0], y0 = y0_param[0];
  __syncthreads();

  const int sample = blockIdx.x * 32 + wv * 16 + lane;  // valid for lane<16

  float y = y0, Yc = 0.f, dYc = 0.f, acc = 0.f;

  // initial Y_and_grad at (t=0, y0)
  if (lane < 16) ystage[wv][lane] = y0;
  {
    float ym = ystage[wv][m];
    input_layer(0.f, ym, sYW0, sYW1, sYbi, hbuf[wv], ubuf[wv], m, g);
    for (int l = 0; l < 3; ++l) hidden_hu(hbuf[wv], ubuf[wv], sYW[l], sYb[l], m, g);
    float2 yd = out_dots(hbuf[wv], ubuf[wv], sYWo, m, g);
    Yc = yd.x + Ybo;
    dYc = yd.y;
  }

#pragma unroll 1
  for (int n = 0; n < N_STEPS; ++n) {
    const float tn = (float)n * DT;

    // ---- q = q_net(t_n, y_n) ----
    if (lane < 16) ystage[wv][lane] = y;
    float ym = ystage[wv][m];
    input_layer(tn, ym, sQW0, sQW1, sQbi, hbuf[wv], nullptr, m, g);
    for (int l = 0; l < 3; ++l) hidden_h(hbuf[wv], sQW[l], sQb[l], m, g);
    float2 qd = out_dots(hbuf[wv], nullptr, sQWo, m, g);
    float q = qd.x + qbo;  // q for sample row m, valid in every lane

    // ---- Euler step + target (lanes 0..15 own sample state) ----
    float y1 = 0.f, Ytil = 0.f;
    if (lane < 16) {
      float dw = dW[n * B_PATHS + sample] * SQRT_DT;
      if (n + 1 < N_STEPS)
        __builtin_prefetch(&dW[(n + 1) * B_PATHS + sample], 0, 3);
      y1 = y + q * DT + SIGMA_C * dw;
      Ytil = Yc - q * q * DT + (SIGMA_C * dYc) * dw;
      ystage[wv][lane] = y1;
    }
    float y1m = ystage[wv][m];

    // ---- (Y_{n+1}, dY_{n+1}) = Y_and_grad(t_{n+1}, y_{n+1}) ----
    const float t1 = tn + DT;
    input_layer(t1, y1m, sYW0, sYW1, sYbi, hbuf[wv], ubuf[wv], m, g);
    for (int l = 0; l < 3; ++l) hidden_hu(hbuf[wv], ubuf[wv], sYW[l], sYb[l], m, g);
    float2 yd = out_dots(hbuf[wv], ubuf[wv], sYWo, m, g);
    float Y1 = yd.x + Ybo, dY1 = yd.y;

    if (lane < 16) {
      float r = Y1 - Ytil;
      acc += r * r;
      y = y1;
      Yc = Y1;
      dYc = dY1;
    }
  }

  // terminal terms: g(y)=y^2, grad=2y
  if (lane < 16) {
    float r1 = Yc - y * y;
    float r2 = dYc - 2.f * y;
    acc += r1 * r1 + r2 * r2;
  }

  // deterministic reduction: wave -> block -> partials[block]
  acc += __shfl_xor(acc, 16);
  acc += __shfl_xor(acc, 8);
  acc += __shfl_xor(acc, 4);
  acc += __shfl_xor(acc, 2);
  acc += __shfl_xor(acc, 1);
  if (lane == 0) wsum[wv] = acc;
  __syncthreads();
  if (tid == 0) partials[blockIdx.x] = wsum[0] + wsum[1];
}

__global__ void reduce_partials(const float* __restrict__ part,
                                float* __restrict__ out, int n, float scale) {
  if (blockIdx.x == 0 && threadIdx.x == 0) {
    float s = 0.f;
    for (int i = 0; i < n; ++i) s += part[i];  // fixed order: deterministic
    out[0] = s * scale;
  }
}

extern "C" void kernel_launch(void* const* d_in, const int* in_sizes, int n_in,
                              void* d_out, int out_size, void* d_ws, size_t ws_size,
                              hipStream_t stream) {
  const float* Y_Win  = (const float*)d_in[0];
  const float* Y_bin  = (const float*)d_in[1];
  const float* Y_Whid = (const float*)d_in[2];
  const float* Y_bhid = (const float*)d_in[3];
  const float* Y_Wout = (const float*)d_in[4];
  const float* Y_bout = (const float*)d_in[5];
  const float* q_Win  = (const float*)d_in[6];
  const float* q_bin  = (const float*)d_in[7];
  const float* q_Whid = (const float*)d_in[8];
  const float* q_bhid = (const float*)d_in[9];
  const float* q_Wout = (const float*)d_in[10];
  const float* q_bout = (const float*)d_in[11];
  const float* y0p    = (const float*)d_in[12];
  const float* dW     = (const float*)d_in[13];
  float* partials = (float*)d_ws;

  const int nblocks = B_PATHS / 32;  // 128 blocks x 2 waves x 16 samples
  fbsnn_kernel<<<nblocks, 64, 0, stream>>>(Y_Win, Y_bin, Y_Whid, Y_bhid, Y_Wout,
                                           Y_bout, q_Win, q_bin, q_Whid, q_bhid,
                                           q_Wout, q_bout, y0p, dW, partials);
  reduce_partials<<<1, 32, 0, stream>>>(partials, (float*)d_out, nblocks,
                                        1.0f / (float)B_PATHS);
}
